// Shapley_Q_33535104647957
// MI455X (gfx1250) — matrix operand order repaired and checked by reading, not compiled, not run on hardware
//
#include <hip/hip_runtime.h>

// Problem constants (from reference): B=512, S=64, N=8, D=64, A=16, H=256
#define BB   512
#define SS   64
#define NN   8
#define DD   64
#define AAi  16
#define HH   256
#define KOBS 512   // N*D  (obs part of x, shared per batch)
#define KACT 128   // N*A  (act part of x, per-row)
#define NT   16    // H/16 column tiles
#define KS1  4     // KACT/32 k-steps for layer1-act
#define KS2  8     // H/32   k-steps for layer2
#define MROWS 32   // rows per block = 4 s-values x 8 players (2 M-tiles)

typedef __attribute__((ext_vector_type(16))) _Float16 v16h;
typedef __attribute__((ext_vector_type(8)))  _Float16 v8h;
typedef __attribute__((ext_vector_type(8)))  float    v8f;

__device__ __forceinline__ float mish_f(float x) {
    // mish(x) = x * tanh(softplus(x)) = x * ((1+e)^2 - 1)/((1+e)^2 + 1), e = exp(x)
    // fast: replace divide with v_rcp_f32 (f16-precision network, rcp error irrelevant)
    float e = __expf(fminf(x, 15.0f));   // clamp: for x>15, ratio == 1 anyway
    float p = 1.0f + e;
    p = p * p;
    return x * (p - 1.0f) * __builtin_amdgcn_rcpf(p + 1.0f);
}

// ---------------- prep kernels ----------------

__global__ void zero_out_kernel(float* __restrict__ out, int n) {
    int i = blockIdx.x * blockDim.x + threadIdx.x;
    if (i < n) out[i] = 0.0f;
}

// Pack a row-major f32 weight matrix W[k][ld] into f16 WMMA B-fragment order:
// dst[((ntile*ksteps + kstep)*32 + lane)*16 + j] = W[krowoff + kstep*32 + (lane>>4)*8 + j + (j>=8 ? 8 : 0)][ntile*16 + (lane&15)]
__global__ void pack_weights_kernel(const float* __restrict__ W, _Float16* __restrict__ dst,
                                    int ksteps, int krowoff, int ld) {
    int idx = blockIdx.x * blockDim.x + threadIdx.x;
    int total = NT * ksteps * 32;
    if (idx >= total) return;
    int lane  = idx & 31;
    int kstep = (idx >> 5) % ksteps;
    int ntile = idx / (32 * ksteps);
    int col = lane & 15;
    int kg  = lane >> 4;
    _Float16* o = dst + ((size_t)(ntile * ksteps + kstep) * 32 + lane) * 16;
    int kbase = krowoff + kstep * 32 + kg * 8;
#pragma unroll
    for (int j = 0; j < 16; ++j)
        o[j] = (_Float16)W[(size_t)(kbase + j + (j >= 8 ? 8 : 0)) * ld + ntile * 16 + col];
}

// h1obs[b][h] = b1[h] + sum_k state_flat[b][k] * W1[k][h]   (obs part, shared by all (s,i) of b)
__global__ void obs_gemm_kernel(const float* __restrict__ state, const float* __restrict__ W1,
                                const float* __restrict__ b1, float* __restrict__ h1obs) {
    int b = blockIdx.x;
    int h = threadIdx.x;
    const float* s = state + (size_t)b * KOBS;
    float acc = b1[h];
#pragma unroll 4
    for (int k = 0; k < KOBS; ++k)
        acc = fmaf(s[k], W1[(size_t)k * HH + h], acc);
    h1obs[(size_t)b * HH + h] = acc;
}

// ---------------- main fused MLP kernel ----------------
// grid = B*16 blocks; block = 512 threads = 16 waves.
// Block (b, sb) handles s in {4*sb .. 4*sb+3} -> 32 M-rows (two 16-row M-tiles).
// Wave w owns N-tile w (16 cols). Each loaded B fragment feeds 2 WMMAs (one per M-tile).

__global__ __launch_bounds__(512, 1) void shapley_main_kernel(
    const float* __restrict__ action,    // (B,N,A) f32
    const int*   __restrict__ perm,      // (B,S,N)
    const _Float16* __restrict__ w1p,    // packed act-part of W1 (B-fragment order)
    const _Float16* __restrict__ w2p,    // packed W2
    const float* __restrict__ h1obs,     // (B,H), includes b1
    const float* __restrict__ b2,        // (H)
    const float* __restrict__ W3,        // (H)
    const float* __restrict__ b3,        // (1)
    float* __restrict__ out)             // 2 * (B,N)
{
    __shared__ int p_lds[MROWS];
    __shared__ __align__(16) _Float16 xact[MROWS][136];   // 128 + 8 pad (bank spread)
    __shared__ __align__(16) _Float16 h1[MROWS][264];     // 256 + 8 pad
    __shared__ float qacc[MROWS];

    const int tid    = threadIdx.x;
    const int lane   = tid & 31;
    const int wv     = tid >> 5;            // wave id == N-tile id
    const int b      = blockIdx.x >> 4;
    const int s_base = (blockIdx.x & 15) * 4;

    if (tid < MROWS) {
        int s_local = tid >> 3, i = tid & 7;
        p_lds[tid] = perm[((size_t)b * SS + s_base + s_local) * NN + i];
        qacc[tid] = 0.0f;
    }
    __syncthreads();

    // Build x_act rows in LDS: act[row][j*A+a] = action[b, perm[b,s,j], a] * (j <= perm[b,s,i])
    for (int idx = tid; idx < MROWS * KACT; idx += 512) {
        int row = idx >> 7, c = idx & 127;
        int j = c >> 4, a = c & 15;
        int s_local = row >> 3, i = row & 7;
        int pi = p_lds[s_local * 8 + i];
        int pj = p_lds[s_local * 8 + j];
        float v = (j <= pi) ? action[((size_t)b * NN + pj) * AAi + a] : 0.0f;
        xact[row][c] = (_Float16)v;
    }
    __syncthreads();

    const int col = wv * 16 + (lane & 15);  // this lane's output column
    const int m   = lane & 15;              // A-fragment row within an M-tile
    const int kg  = lane >> 4;              // K half-group

    // Warm near caches for this wave's W2 fragments while layer 1 runs
    __builtin_prefetch((const char*)(w2p + (size_t)wv * KS2 * 32 * 16), 0, 0);

    // ---- layer 1 (act part), C initialized with precomputed obs part ----
    v8f c1a, c1b;
    {
        float ci = h1obs[(size_t)b * HH + col];   // same for all rows of this b
#pragma unroll
        for (int v = 0; v < 8; ++v) { c1a[v] = ci; c1b[v] = ci; }
    }
    const v16h* bfr1 = (const v16h*)w1p + (size_t)wv * KS1 * 32;
#pragma unroll
    for (int ks = 0; ks < KS1; ++ks) {
        v8h lo0 = *(const v8h*)&xact[m][ks * 32 + kg * 8];
        v8h hi0 = *(const v8h*)&xact[m][ks * 32 + 16 + kg * 8];
        v8h lo1 = *(const v8h*)&xact[m + 16][ks * 32 + kg * 8];
        v8h hi1 = *(const v8h*)&xact[m + 16][ks * 32 + 16 + kg * 8];
        v16h a0, a1;
#pragma unroll
        for (int t = 0; t < 8; ++t) {
            a0[t] = lo0[t]; a0[8 + t] = hi0[t];
            a1[t] = lo1[t]; a1[8 + t] = hi1[t];
        }
        v16h bb = bfr1[ks * 32 + lane];          // loaded once, used twice
        c1a = __builtin_amdgcn_wmma_f32_16x16x32_f16(false, a0, false, bb,
                                                     (short)0, c1a, false, false);
        c1b = __builtin_amdgcn_wmma_f32_16x16x32_f16(false, a1, false, bb,
                                                     (short)0, c1b, false, false);
    }

    // mish + stage h1 (f16) in LDS. C layout: VGPR v -> row v + 8*kg, col fixed.
#pragma unroll
    for (int v = 0; v < 8; ++v) {
        int row = v + (kg << 3);
        h1[row][col]      = (_Float16)mish_f(c1a[v]);
        h1[row + 16][col] = (_Float16)mish_f(c1b[v]);
    }
    __syncthreads();

    // ---- layer 2 ----
    v8f c2a, c2b;
    {
        float ci = b2[col];
#pragma unroll
        for (int v = 0; v < 8; ++v) { c2a[v] = ci; c2b[v] = ci; }
    }
    const v16h* bfr2 = (const v16h*)w2p + (size_t)wv * KS2 * 32;
#pragma unroll
    for (int ks = 0; ks < KS2; ++ks) {
        v8h lo0 = *(const v8h*)&h1[m][ks * 32 + kg * 8];
        v8h hi0 = *(const v8h*)&h1[m][ks * 32 + 16 + kg * 8];
        v8h lo1 = *(const v8h*)&h1[m + 16][ks * 32 + kg * 8];
        v8h hi1 = *(const v8h*)&h1[m + 16][ks * 32 + 16 + kg * 8];
        v16h a0, a1;
#pragma unroll
        for (int t = 0; t < 8; ++t) {
            a0[t] = lo0[t]; a0[8 + t] = hi0[t];
            a1[t] = lo1[t]; a1[8 + t] = hi1[t];
        }
        v16h bb = bfr2[ks * 32 + lane];
        c2a = __builtin_amdgcn_wmma_f32_16x16x32_f16(false, a0, false, bb,
                                                     (short)0, c2a, false, false);
        c2b = __builtin_amdgcn_wmma_f32_16x16x32_f16(false, a1, false, bb,
                                                     (short)0, c2b, false, false);
    }

    // ---- layer 3: q[row] = sum_col mish(h2[row][col]) * W3[col]  (+ b3, /S) ----
    float wt = W3[col];
    float pra[8], prb[8];
#pragma unroll
    for (int v = 0; v < 8; ++v) {
        pra[v] = mish_f(c2a[v]) * wt;
        prb[v] = mish_f(c2b[v]) * wt;
    }
    // reduce across the 16 lanes of each half (cols of this wave)
#pragma unroll
    for (int msk = 1; msk < 16; msk <<= 1) {
#pragma unroll
        for (int v = 0; v < 8; ++v) {
            pra[v] += __shfl_xor(pra[v], msk, 32);
            prb[v] += __shfl_xor(prb[v], msk, 32);
        }
    }
    if ((lane & 15) == 0) {
#pragma unroll
        for (int v = 0; v < 8; ++v) {
            atomicAdd(&qacc[v + (kg << 3)], pra[v]);        // rows 0..15
            atomicAdd(&qacc[16 + v + (kg << 3)], prb[v]);   // rows 16..31
        }
    }
    __syncthreads();

    if (tid < MROWS) {
        float q = (qacc[tid] + b3[0]) * (1.0f / (float)SS);
        int i = tid & 7;
        atomicAdd(&out[b * NN + i], q);                 // q1 (first tuple element)
        atomicAdd(&out[BB * NN + b * NN + i], q);       // q1 (second tuple element)
    }
}

// ---------------- host side ----------------

extern "C" void kernel_launch(void* const* d_in, const int* in_sizes, int n_in,
                              void* d_out, int out_size, void* d_ws, size_t ws_size,
                              hipStream_t stream) {
    (void)in_sizes; (void)n_in; (void)out_size; (void)ws_size;

    const float* state  = (const float*)d_in[0];
    const float* action = (const float*)d_in[1];
    const float* W1     = (const float*)d_in[2];
    const float* b1     = (const float*)d_in[3];
    const float* W2     = (const float*)d_in[4];
    const float* b2     = (const float*)d_in[5];
    const float* W3     = (const float*)d_in[6];
    const float* b3     = (const float*)d_in[7];
    const int*   perm   = (const int*)d_in[8];
    float* out = (float*)d_out;

    char* ws = (char*)d_ws;
    _Float16* w1p  = (_Float16*)ws;                        // 16*4*32*16 f16 = 64 KB
    _Float16* w2p  = (_Float16*)(ws + 65536);              // 16*8*32*16 f16 = 128 KB
    float*    h1o  = (float*)(ws + 65536 + 131072);        // 512*256 f32   = 512 KB

    zero_out_kernel<<<(2 * BB * NN + 255) / 256, 256, 0, stream>>>(out, 2 * BB * NN);
    pack_weights_kernel<<<(NT * KS1 * 32 + 255) / 256, 256, 0, stream>>>(W1, w1p, KS1, KOBS, HH);
    pack_weights_kernel<<<(NT * KS2 * 32 + 255) / 256, 256, 0, stream>>>(W2, w2p, KS2, 0, HH);
    obs_gemm_kernel<<<BB, HH, 0, stream>>>(state, W1, b1, h1o);
    shapley_main_kernel<<<BB * 16, 512, 0, stream>>>(action, perm, w1p, w2p, h1o, b2, W3, b3, out);
}
